// Model_4260607558339
// MI455X (gfx1250) — matrix operand order repaired and checked by reading, not compiled, not run on hardware
//
#include <hip/hip_runtime.h>

// ---------------------------------------------------------------------------
// CDNA5 (gfx1250) pointer-generator forward pass.
// GEMMs use v_wmma_f32_16x16x32_bf16 with fragment-packed B operands:
// per (ntile,ktile) the 32 lanes' 16 bf16 values are contiguous -> one 32B
// vector load per WMMA instead of 16 strided u16 gathers.
// ---------------------------------------------------------------------------

typedef __bf16 bf16_t;
typedef __attribute__((ext_vector_type(16))) __bf16 v16bf;
typedef __attribute__((ext_vector_type(8)))  __bf16 v8bf;
typedef __attribute__((ext_vector_type(8)))  float  v8f;

#define CDIV(a,b) (((a)+(b)-1)/(b))

// ---- model constants -------------------------------------------------------
#define NB   16      // batch
#define NSRC 400     // src length
#define NSEC 4
#define NWL  100
#define NH   256
#define NE   128
#define NV   50000
#define NDT  50
#define NT   49      // decoder steps
#define N4H  1024
#define N2H  512
#define NVX  50050   // V + OOV

// ---- input indices (flattened setup_inputs dict order) ---------------------
enum {
  IN_EMB = 0,
  IN_ENCF_WIH, IN_ENCF_WHH, IN_ENCF_BIH, IN_ENCF_BHH,
  IN_ENCB_WIH, IN_ENCB_WHH, IN_ENCB_BIH, IN_ENCB_BHH,
  IN_SECF_WIH, IN_SECF_WHH, IN_SECF_BIH, IN_SECF_BHH,
  IN_SECB_WIH, IN_SECB_WHH, IN_SECB_BIH, IN_SECB_BHH,
  IN_WSEC, IN_BSEC,
  IN_WFEATENC,
  IN_REDH, IN_BRH, IN_REDC, IN_BRC,
  IN_DEC_WIH, IN_DEC_WHH, IN_DEC_BIH, IN_DEC_BHH,
  IN_WCC, IN_BCC,
  IN_DP, IN_BDP,
  IN_VW, IN_WCOV,
  IN_WD, IN_BWD,
  IN_WFEATSEC, IN_VS,
  IN_WPG, IN_BPG,
  IN_WOUT, IN_BOUT,
  IN_ENC_INPUT, IN_ENC_MASK, IN_SEC_MASK, IN_ENC_OOV, IN_ZEROS_OOV,
  IN_CONTEXT, IN_COVERAGE, IN_DEC_INPUT, IN_DEC_MASK, IN_DEC_LENS, IN_TARGET,
  IN_SEC_NUM, IN_SEC_LEN, IN_DEC_LEN
};

// ---- WMMA fragment lane indexing (ISA 7.12.2, wave32) ----------------------
struct WmmaIdx { int arow, akoff, crow, ccol; };

__device__ __forceinline__ WmmaIdx widx() {
  const int lane = threadIdx.x & 31;
  WmmaIdx w;
  w.arow  = lane & 15;
  w.akoff = (lane >> 4) * 8;   // lanes 0-15: K 0..7 / 16..23; lanes 16-31: 8..15 / 24..31
  w.ccol  = lane & 15;
  w.crow  = (lane >> 4) * 8;   // VGPR g -> M = g (+8 for upper half-wave)
  return w;
}

// A fragment: two contiguous 16B vector loads (row-major A, 16B aligned).
__device__ __forceinline__ v16bf load_a(const bf16_t* A, int lda, const WmmaIdx& w) {
  const bf16_t* p = A + w.arow * lda + w.akoff;
  union { v16bf v; v8bf h[2]; } u;
  u.h[0] = *(const v8bf*)(p);
  u.h[1] = *(const v8bf*)(p + 16);
  return u.v;
}

// One 16x16x32 step with fragment-packed B tile (lane-major, 512 elems/tile).
__device__ __forceinline__ v8f wmma_pk(const bf16_t* A, int lda,
                                       const bf16_t* Btile,
                                       const WmmaIdx& w, v8f acc) {
  v16bf av = load_a(A, lda, w);
  v16bf bv = *(const v16bf*)(Btile + (threadIdx.x & 31) * 16);
  return __builtin_amdgcn_wmma_f32_16x16x32_bf16(false, av, false, bv,
                                                 (short)0, acc, false, false);
}

__device__ __forceinline__ float sigf(float x) { return 1.f / (1.f + expf(-x)); }

// ---------------------------------------------------------------------------
// Weight conversion: W (N x K f32, row-major, torch layout) -> fragment-packed
// bf16 B(K x N).  Tile (nt,kt) at ((nt*(K/32)+kt)*512); lane L holds
// col = nt*16 + (L&15), k = kt*32 + (L>>4)*8 + (j&7) + (j>>3)*16 for j=0..15.
// ---------------------------------------------------------------------------
__global__ void k_cvt_pack(const float* __restrict__ W, bf16_t* __restrict__ P,
                           int N, int K) {
  size_t i = (size_t)blockIdx.x * blockDim.x + threadIdx.x;
  if (i >= (size_t)N * K) return;
  const int Kt = K >> 5;
  size_t tile = i >> 9;
  int r = (int)(i & 511);
  int lane = r >> 4, j = r & 15;
  int kt = (int)(tile % Kt), nt = (int)(tile / Kt);
  int col = nt * 16 + (lane & 15);
  int k = kt * 32 + (lane >> 4) * 8 + (j & 7) + (j >> 3) * 16;
  P[i] = (bf16_t)W[(size_t)col * K + k];
}

__global__ void k_f2b(const float* __restrict__ s, bf16_t* __restrict__ d, size_t n) {
  size_t i = (size_t)blockIdx.x * blockDim.x + threadIdx.x;
  if (i < n) d[i] = (bf16_t)s[i];
}

__global__ void k_embed(const float* __restrict__ emb, const int* __restrict__ tok,
                        bf16_t* __restrict__ out, int rows) {
  size_t i = (size_t)blockIdx.x * blockDim.x + threadIdx.x;
  if (i >= (size_t)rows * NE) return;
  size_t r = i / NE, j = i % NE;
  out[i] = (bf16_t)emb[(size_t)tok[r] * NE + j];
}

__global__ void k_init(const float* __restrict__ ctx_in, const float* __restrict__ cov_in,
                       float* __restrict__ ctx, float* __restrict__ cov,
                       float* __restrict__ loss_acc) {
  int i = blockIdx.x * blockDim.x + threadIdx.x;
  if (i < NB * N2H)  ctx[i] = ctx_in[i];
  if (i < NB * NSRC) cov[i] = cov_in[i];
  if (i < NB)        loss_acc[i] = 0.f;
}

// ---------------------------------------------------------------------------
// Generic WMMA GEMM: C(MxN f32) = A(MxK bf16) @ Bpacked [+b1][+b2][relu]
// block 256 (8 waves), grid = (M/16, ceil(N/128))
// ---------------------------------------------------------------------------
__global__ void k_gemm(const bf16_t* __restrict__ A, const bf16_t* __restrict__ B,
                       float* __restrict__ C, int M, int N, int K,
                       const float* __restrict__ bias1,
                       const float* __restrict__ bias2, int relu) {
  const int wave = threadIdx.x >> 5;
  const int nt = blockIdx.y * 8 + wave;
  if (nt * 16 >= N) return;
  const int mt = blockIdx.x;
  const int Kt = K >> 5;
  WmmaIdx w = widx();
  v8f cf = {};
  const bf16_t* Ar = A + (size_t)mt * 16 * K;
  const bf16_t* Bn = B + (size_t)nt * Kt * 512;
  for (int kt = 0; kt < Kt; ++kt)
    cf = wmma_pk(Ar + kt * 32, K, Bn + (size_t)kt * 512, w, cf);
  const int col = nt * 16 + w.ccol;
  float bia = (bias1 ? bias1[col] : 0.f) + (bias2 ? bias2[col] : 0.f);
#pragma unroll
  for (int g = 0; g < 8; ++g) {
    float v = cf[g] + bia;
    if (relu) v = fmaxf(v, 0.f);
    C[(size_t)(mt * 16 + w.crow + g) * N + col] = v;
  }
}

// ---------------------------------------------------------------------------
// Persistent word-level bi-LSTM: one block per (section, dir), h/c in LDS.
// ---------------------------------------------------------------------------
__global__ void __launch_bounds__(1024)
k_lstm_enc(const float* __restrict__ xprojf, const float* __restrict__ xprojb,
           const bf16_t* __restrict__ whhF, const bf16_t* __restrict__ whhB,
           float* __restrict__ enc_out, float* __restrict__ hs_all) {
  const int sec = blockIdx.x, dir = blockIdx.y;
  const float* xproj = dir ? xprojb : xprojf;
  const bf16_t* whh  = dir ? whhB : whhF;
  __shared__ bf16_t sh_h[NB * NH];     // 8 KB
  __shared__ bf16_t sh_g[NB * N4H];    // 32 KB
  __shared__ float  sh_c[NB * NH];     // 16 KB
  const int tid = threadIdx.x;
  for (int i = tid; i < NB * NH; i += 1024) { sh_h[i] = (bf16_t)0.f; sh_c[i] = 0.f; }
  __syncthreads();
  const int wave = tid >> 5;
  WmmaIdx w = widx();
  for (int t = 0; t < NWL; ++t) {
    const int tg = dir ? (NWL - 1 - t) : t;
    for (int tt = 0; tt < 2; ++tt) {
      const int nt = wave * 2 + tt;
      const int col = nt * 16 + w.ccol;
      const bf16_t* Bn = whh + (size_t)nt * 8 * 512;   // Kt = 256/32 = 8
      v8f cf;
#pragma unroll
      for (int g = 0; g < 8; ++g) {
        const int b = w.crow + g;
        cf[g] = xproj[(size_t)(b * NSRC + sec * NWL + tg) * N4H + col];
      }
#pragma unroll
      for (int kt = 0; kt < 8; ++kt)
        cf = wmma_pk(sh_h + kt * 32, NH, Bn + kt * 512, w, cf);
#pragma unroll
      for (int g = 0; g < 8; ++g)
        sh_g[(w.crow + g) * N4H + col] = (bf16_t)cf[g];
    }
    __syncthreads();
    for (int i = tid; i < NB * NH; i += 1024) {
      const int b = i >> 8, j = i & 255;
      const float gi = (float)sh_g[b * N4H + j];
      const float gf = (float)sh_g[b * N4H + NH + j];
      const float gg = (float)sh_g[b * N4H + 2 * NH + j];
      const float go = (float)sh_g[b * N4H + 3 * NH + j];
      const float cv = sigf(gf) * sh_c[i] + sigf(gi) * tanhf(gg);
      const float hv = sigf(go) * tanhf(cv);
      sh_c[i] = cv;
      sh_h[i] = (bf16_t)hv;
      enc_out[(size_t)(b * NSRC + sec * NWL + tg) * N2H + dir * NH + j] = hv;
    }
    __syncthreads();
  }
  for (int i = tid; i < NB * NH; i += 1024) {
    const int b = i >> 8, j = i & 255;
    hs_all[(size_t)((sec * 2 + dir) * NB + b) * NH + j] = (float)sh_h[i];
  }
}

// ---------------------------------------------------------------------------
// Section-level bi-LSTM (T=4). blockIdx.x = dir. xps rows ordered (b, t).
// ---------------------------------------------------------------------------
__global__ void __launch_bounds__(1024)
k_lstm_sec(const float* __restrict__ xpsf, const float* __restrict__ xpsb,
           const bf16_t* __restrict__ whhF, const bf16_t* __restrict__ whhB,
           float* __restrict__ sec_out, float* __restrict__ hS, float* __restrict__ cS) {
  const int dir = blockIdx.x;
  const float* xps  = dir ? xpsb : xpsf;
  const bf16_t* whh = dir ? whhB : whhF;
  __shared__ bf16_t sh_h[NB * NH];
  __shared__ bf16_t sh_g[NB * N4H];
  __shared__ float  sh_c[NB * NH];
  const int tid = threadIdx.x;
  for (int i = tid; i < NB * NH; i += 1024) { sh_h[i] = (bf16_t)0.f; sh_c[i] = 0.f; }
  __syncthreads();
  const int wave = tid >> 5;
  WmmaIdx w = widx();
  for (int t = 0; t < NSEC; ++t) {
    const int tg = dir ? (NSEC - 1 - t) : t;
    for (int tt = 0; tt < 2; ++tt) {
      const int nt = wave * 2 + tt;
      const int col = nt * 16 + w.ccol;
      const bf16_t* Bn = whh + (size_t)nt * 8 * 512;
      v8f cf;
#pragma unroll
      for (int g = 0; g < 8; ++g)
        cf[g] = xps[(size_t)((w.crow + g) * NSEC + tg) * N4H + col];
#pragma unroll
      for (int kt = 0; kt < 8; ++kt)
        cf = wmma_pk(sh_h + kt * 32, NH, Bn + kt * 512, w, cf);
#pragma unroll
      for (int g = 0; g < 8; ++g)
        sh_g[(w.crow + g) * N4H + col] = (bf16_t)cf[g];
    }
    __syncthreads();
    for (int i = tid; i < NB * NH; i += 1024) {
      const int b = i >> 8, j = i & 255;
      const float gi = (float)sh_g[b * N4H + j];
      const float gf = (float)sh_g[b * N4H + NH + j];
      const float gg = (float)sh_g[b * N4H + 2 * NH + j];
      const float go = (float)sh_g[b * N4H + 3 * NH + j];
      const float cv = sigf(gf) * sh_c[i] + sigf(gi) * tanhf(gg);
      const float hv = sigf(go) * tanhf(cv);
      sh_c[i] = cv;
      sh_h[i] = (bf16_t)hv;
      sec_out[(size_t)(b * NSEC + tg) * N2H + dir * NH + j] = hv;
    }
    __syncthreads();
  }
  for (int i = tid; i < NB * NH; i += 1024) {
    const int b = i >> 8, j = i & 255;
    hS[(size_t)(dir * NB + b) * NH + j] = (float)sh_h[i];
    cS[(size_t)(dir * NB + b) * NH + j] = sh_c[i];
  }
}

// sec_rep (s*16+b rows) -> secin_bf ordered (b, s)
__global__ void k_perm_secin(const float* __restrict__ sec_rep, bf16_t* __restrict__ out) {
  int i = blockIdx.x * blockDim.x + threadIdx.x;
  if (i >= 64 * NH) return;
  int row = i >> 8, k = i & 255;
  int bp = row >> 2, s = row & 3;
  out[i] = (bf16_t)sec_rep[(s * NB + bp) * NH + k];
}

// hS/cS (2,16,256) -> (16,512) bf16 concat
__global__ void k_hs_cat(const float* __restrict__ hS, const float* __restrict__ cS,
                         bf16_t* __restrict__ hcat, bf16_t* __restrict__ ccat) {
  int i = blockIdx.x * blockDim.x + threadIdx.x;
  if (i >= NB * N2H) return;
  int b = i >> 9, k = i & 511;
  int d = k >> 8, j = k & 255;
  hcat[i] = (bf16_t)hS[(d * NB + b) * NH + j];
  ccat[i] = (bf16_t)cS[(d * NB + b) * NH + j];
}

// ---------------------------------------------------------------------------
// Decoder step kernels
// ---------------------------------------------------------------------------

// x = [ctx, emb(tok)] @ w_cc^T + b_cc.  block 256 (8 waves = 8 N-tiles)
__global__ void k_dec_x(const float* __restrict__ emb, const int* __restrict__ dec_in,
                        int t, const float* __restrict__ ctx,
                        const bf16_t* __restrict__ wccB, const float* __restrict__ b_cc,
                        float* __restrict__ x_f, bf16_t* __restrict__ x_bf) {
  __shared__ bf16_t shA[NB * 640];   // 20 KB
  const int tid = threadIdx.x;
  for (int i = tid; i < NB * 640; i += 256) {
    int b = i / 640, k = i % 640;
    float v = (k < N2H) ? ctx[b * N2H + k]
                        : emb[(size_t)dec_in[b * NDT + t] * NE + (k - N2H)];
    shA[i] = (bf16_t)v;
  }
  __syncthreads();
  const int wave = tid >> 5;
  WmmaIdx w = widx();
  const bf16_t* Bn = wccB + (size_t)wave * 20 * 512;   // Kt = 640/32 = 20
  v8f cf = {};
  for (int kt = 0; kt < 20; ++kt)
    cf = wmma_pk(shA + kt * 32, 640, Bn + (size_t)kt * 512, w, cf);
  const int col = wave * 16 + w.ccol;
#pragma unroll
  for (int g = 0; g < 8; ++g) {
    float v = cf[g] + b_cc[col];
    x_f [(w.crow + g) * NE + col] = v;
    x_bf[(w.crow + g) * NE + col] = (bf16_t)v;
  }
}

// decoder LSTM step: gates = x@Wih^T + h@Whh^T + bih + bhh
__global__ void __launch_bounds__(1024)
k_dec_lstm(const bf16_t* __restrict__ x_bf,
           const bf16_t* __restrict__ wihB, const bf16_t* __restrict__ whhB,
           const float* __restrict__ bih, const float* __restrict__ bhh,
           float* __restrict__ h, float* __restrict__ c,
           float* __restrict__ ds_f, bf16_t* __restrict__ ds_bf) {
  __shared__ bf16_t sh_h[NB * NH];
  __shared__ bf16_t sh_g[NB * N4H];
  const int tid = threadIdx.x;
  for (int i = tid; i < NB * NH; i += 1024) sh_h[i] = (bf16_t)h[i];
  __syncthreads();
  const int wave = tid >> 5;
  WmmaIdx w = widx();
  for (int tt = 0; tt < 2; ++tt) {
    const int nt = wave * 2 + tt;
    const int col = nt * 16 + w.ccol;
    const float bia = bih[col] + bhh[col];
    const bf16_t* Bi = wihB + (size_t)nt * 4 * 512;    // Kt = 128/32 = 4
    const bf16_t* Bh = whhB + (size_t)nt * 8 * 512;    // Kt = 256/32 = 8
    v8f cf;
#pragma unroll
    for (int g = 0; g < 8; ++g) cf[g] = bia;
#pragma unroll
    for (int kt = 0; kt < 4; ++kt)
      cf = wmma_pk(x_bf + kt * 32, NE, Bi + kt * 512, w, cf);
#pragma unroll
    for (int kt = 0; kt < 8; ++kt)
      cf = wmma_pk(sh_h + kt * 32, NH, Bh + kt * 512, w, cf);
#pragma unroll
    for (int g = 0; g < 8; ++g)
      sh_g[(w.crow + g) * N4H + col] = (bf16_t)cf[g];
  }
  __syncthreads();
  for (int i = tid; i < NB * NH; i += 1024) {
    const int b = i >> 8, j = i & 255;
    const float gi = (float)sh_g[b * N4H + j];
    const float gf = (float)sh_g[b * N4H + NH + j];
    const float gg = (float)sh_g[b * N4H + 2 * NH + j];
    const float go = (float)sh_g[b * N4H + 3 * NH + j];
    const float cv = sigf(gf) * c[i] + sigf(gi) * tanhf(gg);
    const float hv = sigf(go) * tanhf(cv);
    h[i] = hv; c[i] = cv;
    ds_f [b * N2H + j]      = hv;
    ds_f [b * N2H + NH + j] = cv;
    ds_bf[b * N2H + j]      = (bf16_t)hv;
    ds_bf[b * N2H + NH + j] = (bf16_t)cv;
  }
}

// word attention + coverage + context. block 512 (16 waves).
__global__ void __launch_bounds__(512)
k_attn(const bf16_t* __restrict__ ds_bf, const bf16_t* __restrict__ dpB,
       const float* __restrict__ b_dp, const float* __restrict__ enc_feat,
       const float* __restrict__ enc_out, const float* __restrict__ v_w,
       const float* __restrict__ w_cov, const float* __restrict__ enc_mask,
       float* __restrict__ cov, float* __restrict__ a_w,
       float* __restrict__ ctx, float* __restrict__ covloss) {
  __shared__ float sh_fea[NB * N2H];   // 32 KB
  __shared__ float sh_a[NB * NSRC];    // 25.6 KB
  const int tid = threadIdx.x;
  // dec_fea = ds @ dp^T + b_dp  (32 tiles, 16 waves x 2; Kt = 512/32 = 16)
  {
    const int wave = tid >> 5;
    WmmaIdx w = widx();
    for (int tt = 0; tt < 2; ++tt) {
      const int nt = wave * 2 + tt;
      const int col = nt * 16 + w.ccol;
      const bf16_t* Bn = dpB + (size_t)nt * 16 * 512;
      v8f cf = {};
#pragma unroll
      for (int kt = 0; kt < 16; ++kt)
        cf = wmma_pk(ds_bf + kt * 32, N2H, Bn + (size_t)kt * 512, w, cf);
#pragma unroll
      for (int g = 0; g < 8; ++g)
        sh_fea[(w.crow + g) * N2H + col] = cf[g] + b_dp[col];
    }
  }
  __syncthreads();
  // additive scores with coverage
  for (int p = tid; p < NB * NSRC; p += 512) {
    const int b = p / NSRC, s = p % NSRC;
    const float* ef = enc_feat + (size_t)(b * NSRC + s) * N2H;
    const float cv = cov[b * NSRC + s];
    float acc = 0.f;
    for (int k = 0; k < N2H; ++k)
      acc += tanhf(ef[k] + sh_fea[b * N2H + k] + cv * w_cov[k]) * v_w[k];
    sh_a[p] = acc;
  }
  __syncthreads();
  // masked softmax + coverage loss (pre-update cov) + cov update
  if (tid < NB) {
    const int b = tid;
    float mx = -3.4e38f;
    for (int s = 0; s < NSRC; ++s) mx = fmaxf(mx, sh_a[b * NSRC + s]);
    float sum = 0.f;
    for (int s = 0; s < NSRC; ++s) {
      float e = expf(sh_a[b * NSRC + s] - mx) * enc_mask[b * NSRC + s];
      sh_a[b * NSRC + s] = e; sum += e;
    }
    const float inv = 1.f / sum;
    float cl = 0.f;
    for (int s = 0; s < NSRC; ++s) {
      float av = sh_a[b * NSRC + s] * inv;
      sh_a[b * NSRC + s] = av;
      a_w[b * NSRC + s]  = av;
      cl += fminf(av, cov[b * NSRC + s]);
    }
    covloss[b] = cl;
    for (int s = 0; s < NSRC; ++s) cov[b * NSRC + s] += sh_a[b * NSRC + s];
  }
  __syncthreads();
  // ctx_new = a @ enc_outputs
  for (int i = tid; i < NB * N2H; i += 512) {
    const int b = i >> 9, k = i & 511;
    float acc = 0.f;
    for (int s = 0; s < NSRC; ++s)
      acc += sh_a[b * NSRC + s] * enc_out[(size_t)(b * NSRC + s) * N2H + k];
    ctx[i] = acc;
  }
}

// hctx = [h, ctx_new] bf16 (16 x 768)
__global__ void k_catA(const float* __restrict__ h, const float* __restrict__ ctx,
                       bf16_t* __restrict__ hctx) {
  int i = blockIdx.x * blockDim.x + threadIdx.x;
  if (i >= NB * 768) return;
  int b = i / 768, k = i % 768;
  hctx[i] = (bf16_t)((k < NH) ? h[b * NH + k] : ctx[b * N2H + (k - NH)]);
}

// vocab projection: logits(16x50000) = hctx(16x768) @ woutPacked + b_out
// block 256 (8 waves = 8 N-tiles). Packed B tiles stream straight from L2
// (76.8 MB bf16 fits the 192 MB L2); A resident in LDS.
__global__ void __launch_bounds__(256)
k_vocab(const bf16_t* __restrict__ A, const bf16_t* __restrict__ B,
        const float* __restrict__ bias, float* __restrict__ C) {
  __shared__ bf16_t shA[NB * 768];   // 24 KB
  const int tid = threadIdx.x;
  const int wave = tid >> 5;
  for (int i = tid; i < NB * 768; i += 256) shA[i] = A[i];
  __syncthreads();
  const int nt = blockIdx.x * 8 + wave;        // 0 .. 3127 (3125 valid)
  const int ntc = (nt < NV / 16) ? nt : (NV / 16 - 1);
  const bf16_t* Bn = B + (size_t)ntc * 24 * 512;   // Kt = 768/32 = 24
  WmmaIdx w = widx();
  v8f cf = {};
#pragma unroll 4
  for (int kt = 0; kt < 24; ++kt) {
    if (kt + 1 < 24)
      __builtin_prefetch(Bn + (size_t)(kt + 1) * 512 + (threadIdx.x & 31) * 16, 0, 1);
    cf = wmma_pk(shA + kt * 32, 768, Bn + (size_t)kt * 512, w, cf);
  }
  if (nt < NV / 16) {
    const int ng = nt * 16 + w.ccol;
#pragma unroll
    for (int g = 0; g < 8; ++g)
      C[(size_t)(w.crow + g) * NV + ng] = cf[g] + bias[ng];
  }
}

__global__ void k_rowstats(const float* __restrict__ logits,
                           float* __restrict__ rowmax, float* __restrict__ rowsum) {
  __shared__ float red[256];
  __shared__ float smx;
  const int b = blockIdx.x, tid = threadIdx.x;
  const float* row = logits + (size_t)b * NV;
  float mx = -3.4e38f;
  for (int j = tid; j < NV; j += 256) mx = fmaxf(mx, row[j]);
  red[tid] = mx; __syncthreads();
  for (int s = 128; s > 0; s >>= 1) { if (tid < s) red[tid] = fmaxf(red[tid], red[tid + s]); __syncthreads(); }
  if (tid == 0) smx = red[0];
  __syncthreads();
  float sm = 0.f;
  for (int j = tid; j < NV; j += 256) sm += expf(row[j] - smx);
  red[tid] = sm; __syncthreads();
  for (int s = 128; s > 0; s >>= 1) { if (tid < s) red[tid] += red[tid + s]; __syncthreads(); }
  if (tid == 0) { rowmax[b] = smx; rowsum[b] = red[0]; }
}

__global__ void k_pg(const float* __restrict__ ctx, const float* __restrict__ ds,
                     const float* __restrict__ x, const float* __restrict__ w_pg,
                     const float* __restrict__ b_pg, float* __restrict__ pg) {
  const int b = threadIdx.x;
  if (b >= NB) return;
  float acc = b_pg[0];
  for (int k = 0; k < N2H; ++k) acc += w_pg[k]         * ctx[b * N2H + k];
  for (int k = 0; k < N2H; ++k) acc += w_pg[N2H + k]   * ds [b * N2H + k];
  for (int k = 0; k < NE;  ++k) acc += w_pg[2*N2H + k] * x  [b * NE  + k];
  pg[b] = sigf(acc);
}

__global__ void k_final(const float* __restrict__ logits, const float* __restrict__ rowmax,
                        const float* __restrict__ rowsum, const float* __restrict__ pg,
                        float* __restrict__ fin) {
  size_t i = (size_t)blockIdx.x * blockDim.x + threadIdx.x;
  if (i >= (size_t)NB * NVX) return;
  const int b = (int)(i / NVX), j = (int)(i % NVX);
  fin[i] = (j < NV) ? pg[b] * expf(logits[(size_t)b * NV + j] - rowmax[b]) / rowsum[b] : 0.f;
}

__global__ void k_scatter(const int* __restrict__ oov, const float* __restrict__ a_w,
                          const float* __restrict__ pg, float* __restrict__ fin) {
  int i = blockIdx.x * blockDim.x + threadIdx.x;
  if (i >= NB * NSRC) return;
  const int b = i / NSRC;
  atomicAdd(&fin[(size_t)b * NVX + oov[i]], (1.f - pg[b]) * a_w[i]);
}

__global__ void k_loss_step(const float* __restrict__ fin, const int* __restrict__ target,
                            const float* __restrict__ dec_mask,
                            const float* __restrict__ covloss, int t,
                            float* __restrict__ loss_acc) {
  const int b = threadIdx.x;
  if (b >= NB) return;
  const float tp = fin[(size_t)b * NVX + target[b * NDT + t]] + 1e-12f;
  float lt = -logf(tp) + 1.0f * covloss[b];
  lt *= dec_mask[b * NDT + t];
  loss_acc[b] += lt;
}

__global__ void k_argmax(const float* __restrict__ row, float* __restrict__ outp) {
  __shared__ float bv[256];
  __shared__ int   bi[256];
  const int tid = threadIdx.x;
  float best = -3.4e38f; int bidx = 0;
  for (int j = tid; j < NVX; j += 256) {
    float v = row[j];
    if (v > best) { best = v; bidx = j; }
  }
  bv[tid] = best; bi[tid] = bidx;
  __syncthreads();
  for (int s = 128; s > 0; s >>= 1) {
    if (tid < s && bv[tid + s] > bv[tid]) { bv[tid] = bv[tid + s]; bi[tid] = bi[tid + s]; }
    __syncthreads();
  }
  if (tid == 0) outp[0] = (float)bi[0];
}

__global__ void k_loss_final(const float* __restrict__ loss_acc,
                             const float* __restrict__ dec_lens, float* __restrict__ out) {
  if (threadIdx.x == 0) {
    float s = 0.f;
    for (int b = 0; b < NB; ++b) s += loss_acc[b] / dec_lens[b];
    out[0] = s / (float)NB;
  }
}

// ---------------------------------------------------------------------------
// Host orchestration
// ---------------------------------------------------------------------------
extern "C" void kernel_launch(void* const* d_in, const int* in_sizes, int n_in,
                              void* d_out, int out_size, void* d_ws, size_t ws_size,
                              hipStream_t stream) {
  (void)in_sizes; (void)n_in; (void)out_size; (void)ws_size;
  const float* p_emb   = (const float*)d_in[IN_EMB];
  const float* p_bout  = (const float*)d_in[IN_BOUT];
  const int*   enc_in  = (const int*)d_in[IN_ENC_INPUT];
  const int*   enc_oov = (const int*)d_in[IN_ENC_OOV];
  const int*   dec_in  = (const int*)d_in[IN_DEC_INPUT];
  const int*   target  = (const int*)d_in[IN_TARGET];
  float* out = (float*)d_out;

  char* base = (char*)d_ws;
  size_t off = 0;
  auto alloc = [&](size_t bytes) -> void* {
    void* p = base + off;
    off = (off + bytes + 255) & ~(size_t)255;
    return p;
  };

  // fragment-packed bf16 weights
  bf16_t* embx    = (bf16_t*)alloc((size_t)NB * NSRC * NE * 2);
  bf16_t* wihfB   = (bf16_t*)alloc((size_t)NE  * N4H * 2);
  bf16_t* whhfB   = (bf16_t*)alloc((size_t)NH  * N4H * 2);
  bf16_t* wihbB   = (bf16_t*)alloc((size_t)NE  * N4H * 2);
  bf16_t* whhbB   = (bf16_t*)alloc((size_t)NH  * N4H * 2);
  bf16_t* swihfB  = (bf16_t*)alloc((size_t)NH  * N4H * 2);
  bf16_t* swhhfB  = (bf16_t*)alloc((size_t)NH  * N4H * 2);
  bf16_t* swihbB  = (bf16_t*)alloc((size_t)NH  * N4H * 2);
  bf16_t* swhhbB  = (bf16_t*)alloc((size_t)NH  * N4H * 2);
  bf16_t* wsecB   = (bf16_t*)alloc((size_t)N2H * NH  * 2);
  bf16_t* wfeB    = (bf16_t*)alloc((size_t)N2H * N2H * 2);
  bf16_t* redhB   = (bf16_t*)alloc((size_t)N2H * NH  * 2);
  bf16_t* redcB   = (bf16_t*)alloc((size_t)N2H * NH  * 2);
  bf16_t* dwihB   = (bf16_t*)alloc((size_t)NE  * N4H * 2);
  bf16_t* dwhhB   = (bf16_t*)alloc((size_t)NH  * N4H * 2);
  bf16_t* wccB    = (bf16_t*)alloc((size_t)640 * NE  * 2);
  bf16_t* dpB     = (bf16_t*)alloc((size_t)N2H * N2H * 2);
  bf16_t* woutB   = (bf16_t*)alloc((size_t)768 * NV  * 2);
  // activations
  float*  xprojf  = (float*)alloc((size_t)NB * NSRC * N4H * 4);
  float*  xprojb  = (float*)alloc((size_t)NB * NSRC * N4H * 4);
  float*  encout  = (float*)alloc((size_t)NB * NSRC * N2H * 4);
  bf16_t* encoutb = (bf16_t*)alloc((size_t)NB * NSRC * N2H * 2);
  float*  encfeat = (float*)alloc((size_t)NB * NSRC * N2H * 4);
  float*  hs_all  = (float*)alloc((size_t)NSEC * 2 * NB * NH * 4);
  bf16_t* hs_bf   = (bf16_t*)alloc((size_t)NSEC * 2 * NB * NH * 2);
  float*  sec_rep = (float*)alloc((size_t)64 * NH * 4);
  bf16_t* secinb  = (bf16_t*)alloc((size_t)64 * NH * 2);
  float*  xpsf    = (float*)alloc((size_t)64 * N4H * 4);
  float*  xpsb    = (float*)alloc((size_t)64 * N4H * 4);
  float*  sec_out = (float*)alloc((size_t)NB * NSEC * N2H * 4);
  float*  hS      = (float*)alloc((size_t)2 * NB * NH * 4);
  float*  cS      = (float*)alloc((size_t)2 * NB * NH * 4);
  bf16_t* hScat   = (bf16_t*)alloc((size_t)NB * N2H * 2);
  bf16_t* cScat   = (bf16_t*)alloc((size_t)NB * N2H * 2);
  float*  hst     = (float*)alloc((size_t)NB * NH * 4);
  float*  cst     = (float*)alloc((size_t)NB * NH * 4);
  float*  ctx     = (float*)alloc((size_t)NB * N2H * 4);
  float*  cov     = (float*)alloc((size_t)NB * NSRC * 4);
  float*  x_f     = (float*)alloc((size_t)NB * NE * 4);
  bf16_t* x_bf    = (bf16_t*)alloc((size_t)NB * NE * 2);
  float*  ds_f    = (float*)alloc((size_t)NB * N2H * 4);
  bf16_t* ds_bf   = (bf16_t*)alloc((size_t)NB * N2H * 2);
  float*  a_w     = (float*)alloc((size_t)NB * NSRC * 4);
  float*  covls   = (float*)alloc((size_t)NB * 4);
  bf16_t* hctx    = (bf16_t*)alloc((size_t)NB * 768 * 2);
  float*  logits  = (float*)alloc((size_t)NB * NV * 4);
  float*  rowmax  = (float*)alloc((size_t)NB * 4);
  float*  rowsum  = (float*)alloc((size_t)NB * 4);
  float*  pgv     = (float*)alloc((size_t)NB * 4);
  float*  fin     = (float*)alloc((size_t)NB * NVX * 4);
  float*  lossacc = (float*)alloc((size_t)NB * 4);

  // -- per-call init --------------------------------------------------------
  k_init<<<32, 256, 0, stream>>>((const float*)d_in[IN_CONTEXT],
                                 (const float*)d_in[IN_COVERAGE], ctx, cov, lossacc);

  // -- weight conversions (fragment-packed) ---------------------------------
  auto cvt = [&](int idx, bf16_t* dst, int N, int K) {
    size_t tot = (size_t)N * K;
    k_cvt_pack<<<(unsigned)CDIV(tot, 256), 256, 0, stream>>>((const float*)d_in[idx], dst, N, K);
  };
  cvt(IN_ENCF_WIH, wihfB, N4H, NE);   cvt(IN_ENCF_WHH, whhfB, N4H, NH);
  cvt(IN_ENCB_WIH, wihbB, N4H, NE);   cvt(IN_ENCB_WHH, whhbB, N4H, NH);
  cvt(IN_SECF_WIH, swihfB, N4H, NH);  cvt(IN_SECF_WHH, swhhfB, N4H, NH);
  cvt(IN_SECB_WIH, swihbB, N4H, NH);  cvt(IN_SECB_WHH, swhhbB, N4H, NH);
  cvt(IN_WSEC, wsecB, NH, N2H);
  cvt(IN_WFEATENC, wfeB, N2H, N2H);
  cvt(IN_REDH, redhB, NH, N2H);       cvt(IN_REDC, redcB, NH, N2H);
  cvt(IN_DEC_WIH, dwihB, N4H, NE);    cvt(IN_DEC_WHH, dwhhB, N4H, NH);
  cvt(IN_WCC, wccB, NE, 640);
  cvt(IN_DP, dpB, N2H, N2H);
  cvt(IN_WOUT, woutB, NV, 768);

  // -- encoder --------------------------------------------------------------
  k_embed<<<CDIV(NB * NSRC * NE, 256), 256, 0, stream>>>(p_emb, enc_in, embx, NB * NSRC);
  k_gemm<<<dim3(NB * NSRC / 16, 8), 256, 0, stream>>>(
      embx, wihfB, xprojf, NB * NSRC, N4H, NE,
      (const float*)d_in[IN_ENCF_BIH], (const float*)d_in[IN_ENCF_BHH], 0);
  k_gemm<<<dim3(NB * NSRC / 16, 8), 256, 0, stream>>>(
      embx, wihbB, xprojb, NB * NSRC, N4H, NE,
      (const float*)d_in[IN_ENCB_BIH], (const float*)d_in[IN_ENCB_BHH], 0);
  k_lstm_enc<<<dim3(NSEC, 2), 1024, 0, stream>>>(xprojf, xprojb, whhfB, whhbB,
                                                 encout, hs_all);

  // -- section level --------------------------------------------------------
  k_f2b<<<CDIV(NSEC * 2 * NB * NH, 256), 256, 0, stream>>>(hs_all, hs_bf,
                                                           (size_t)NSEC * 2 * NB * NH);
  k_gemm<<<dim3(4, 2), 256, 0, stream>>>(hs_bf, wsecB, sec_rep, 64, NH, N2H,
                                         (const float*)d_in[IN_BSEC], nullptr, 0);
  k_perm_secin<<<CDIV(64 * NH, 256), 256, 0, stream>>>(sec_rep, secinb);
  k_gemm<<<dim3(4, 8), 256, 0, stream>>>(secinb, swihfB, xpsf, 64, N4H, NH,
      (const float*)d_in[IN_SECF_BIH], (const float*)d_in[IN_SECF_BHH], 0);
  k_gemm<<<dim3(4, 8), 256, 0, stream>>>(secinb, swihbB, xpsb, 64, N4H, NH,
      (const float*)d_in[IN_SECB_BIH], (const float*)d_in[IN_SECB_BHH], 0);
  k_lstm_sec<<<2, 1024, 0, stream>>>(xpsf, xpsb, swhhfB, swhhbB, sec_out, hS, cS);
  k_hs_cat<<<CDIV(NB * N2H, 256), 256, 0, stream>>>(hS, cS, hScat, cScat);
  k_gemm<<<dim3(1, 2), 256, 0, stream>>>(hScat, redhB, hst, NB, NH, N2H,
                                         (const float*)d_in[IN_BRH], nullptr, 1);
  k_gemm<<<dim3(1, 2), 256, 0, stream>>>(cScat, redcB, cst, NB, NH, N2H,
                                         (const float*)d_in[IN_BRC], nullptr, 1);

  // -- enc_feature ----------------------------------------------------------
  k_f2b<<<CDIV(NB * NSRC * N2H, 256), 256, 0, stream>>>(encout, encoutb,
                                                        (size_t)NB * NSRC * N2H);
  k_gemm<<<dim3(NB * NSRC / 16, 4), 256, 0, stream>>>(encoutb, wfeB, encfeat,
                                                      NB * NSRC, N2H, N2H,
                                                      nullptr, nullptr, 0);

  // -- decoder (teacher forced) --------------------------------------------
  for (int t = 0; t < NT; ++t) {
    k_dec_x<<<1, 256, 0, stream>>>(p_emb, dec_in, t, ctx, wccB,
                                   (const float*)d_in[IN_BCC], x_f, x_bf);
    k_dec_lstm<<<1, 1024, 0, stream>>>(x_bf, dwihB, dwhhB,
                                       (const float*)d_in[IN_DEC_BIH],
                                       (const float*)d_in[IN_DEC_BHH],
                                       hst, cst, ds_f, ds_bf);
    k_attn<<<1, 512, 0, stream>>>(ds_bf, dpB, (const float*)d_in[IN_BDP],
                                  encfeat, encout,
                                  (const float*)d_in[IN_VW],
                                  (const float*)d_in[IN_WCOV],
                                  (const float*)d_in[IN_ENC_MASK],
                                  cov, a_w, ctx, covls);
    k_catA<<<CDIV(NB * 768, 256), 256, 0, stream>>>(hst, ctx, hctx);
    k_vocab<<<CDIV(NV / 16, 8), 256, 0, stream>>>(hctx, woutB, p_bout, logits);
    k_rowstats<<<NB, 256, 0, stream>>>(logits, rowmax, rowsum);
    k_pg<<<1, 32, 0, stream>>>(ctx, ds_f, x_f, (const float*)d_in[IN_WPG],
                               (const float*)d_in[IN_BPG], pgv);
    k_final<<<CDIV((size_t)NB * NVX, 256), 256, 0, stream>>>(logits, rowmax, rowsum,
                                                             pgv, fin);
    k_scatter<<<CDIV(NB * NSRC, 256), 256, 0, stream>>>(enc_oov, a_w, pgv, fin);
    k_loss_step<<<1, 32, 0, stream>>>(fin, target, (const float*)d_in[IN_DEC_MASK],
                                      covls, t, lossacc);
    k_argmax<<<1, 256, 0, stream>>>(fin, out + 1 + t);
  }
  k_loss_final<<<1, 32, 0, stream>>>(lossacc, (const float*)d_in[IN_DEC_LENS], out);
}
